// GraphTransformerPyG_74122545594816
// MI455X (gfx1250) — matrix operand order repaired and checked by reading, not compiled, not run on hardware
//
#include <hip/hip_runtime.h>
#include <cstdint>
#include <cstddef>

// ---------------------------------------------------------------------------
// GraphTransformer (PyG TransformerConv x2) forward for MI455X / gfx1250.
//  - Dense node GEMMs: v_wmma_f32_16x16x32_f16, f32 accumulate.
//  - Weights pre-packed once to the LDS image (f16, transposed, padded stride),
//    staged into LDS with global_load_async_to_lds_b128 (ASYNCcnt path).
//  - Edge phase: wave32-per-edge gather/scatter, float4 traffic, shfl reduce.
// ---------------------------------------------------------------------------

typedef __attribute__((ext_vector_type(16))) _Float16 v16h;
typedef __attribute__((ext_vector_type(8)))  _Float16 v8h;
typedef __attribute__((ext_vector_type(8)))  float    v8f;

#define HIDDIM 128
#define NHEADS 4
#define CDIM   32
#define LDK    136                 // padded f16 row stride (272B) -> conflict-free b128
#define WSLOT  (HIDDIM * LDK)      // halfs per packed weight image (17408 = 34816 B)

// order-preserving float<->int encoding so segment-max can use atomicMax(int*)
__device__ __forceinline__ int f2ord(float f) {
    int i = __float_as_int(f);
    return (i >= 0) ? i : (i ^ 0x7fffffff);
}
__device__ __forceinline__ float ord2f(int i) {
    int j = (i >= 0) ? i : (i ^ 0x7fffffff);
    return __int_as_float(j);
}

// ---------------------------------------------------------------------------
// Stage a contiguous pre-packed f16 image from global into LDS using the
// CDNA5 async-to-LDS data mover, then wait ASYNCcnt==0 and barrier.
// ---------------------------------------------------------------------------
__device__ __forceinline__ void async_stage(const _Float16* __restrict__ src,
                                            _Float16* smem, unsigned bytes,
                                            int tid, int nthreads)
{
    const unsigned ldsBase = (unsigned)(size_t)(void*)smem;   // LDS byte offset
    const char* gsrc = (const char*)src;
    for (unsigned off = (unsigned)tid * 16u; off < bytes; off += (unsigned)nthreads * 16u) {
        unsigned laddr = ldsBase + off;
        unsigned long long gaddr = (unsigned long long)(size_t)(gsrc + off);
        asm volatile("global_load_async_to_lds_b128 %0, %1, off"
                     :: "v"(laddr), "v"(gaddr) : "memory");
    }
    asm volatile("s_wait_asynccnt 0" ::: "memory");
    __syncthreads();
}

// ---------------------------------------------------------------------------
// one-time weight pack: W[K][N] f32 row-major -> Wt[N][LDK] f16 (LDS image)
// ---------------------------------------------------------------------------
__global__ void __launch_bounds__(256)
pack_weight(const float* __restrict__ W, _Float16* __restrict__ out)
{
    int t = blockIdx.x * 256 + threadIdx.x;
    if (t >= HIDDIM * HIDDIM) return;
    int kk = t >> 7;
    int nn = t & 127;
    out[nn * LDK + kk] = (_Float16)W[t];
}

// ---------------------------------------------------------------------------
// per-wave helpers: A-fragment load (ISA 7.12.2 16-bit A layout) and the eight
// 16x16 output tiles of a 16x128 strip (K=128) with bias init + bounds-aware store.
// ---------------------------------------------------------------------------
__device__ __forceinline__ void load_afrag(const float* __restrict__ H, int rowBase,
                                           int mrow, int hs, int nrows, v16h afrag[4])
{
    int r = rowBase + mrow;
    if (r >= nrows) r = nrows - 1;              // clamp; stores are guarded
    const float* hrow = H + (size_t)r * HIDDIM;
    #pragma unroll
    for (int kc = 0; kc < 4; ++kc) {
        const int b0 = 32 * kc + 8 * hs;        // half p: K = 32kc+16(p>>3)+8hs+(p&7)
        float4 x0 = *(const float4*)(hrow + b0);
        float4 x1 = *(const float4*)(hrow + b0 + 4);
        float4 y0 = *(const float4*)(hrow + b0 + 16);
        float4 y1 = *(const float4*)(hrow + b0 + 20);
        v16h a;
        a[0]  = (_Float16)x0.x; a[1]  = (_Float16)x0.y;
        a[2]  = (_Float16)x0.z; a[3]  = (_Float16)x0.w;
        a[4]  = (_Float16)x1.x; a[5]  = (_Float16)x1.y;
        a[6]  = (_Float16)x1.z; a[7]  = (_Float16)x1.w;
        a[8]  = (_Float16)y0.x; a[9]  = (_Float16)y0.y;
        a[10] = (_Float16)y0.z; a[11] = (_Float16)y0.w;
        a[12] = (_Float16)y1.x; a[13] = (_Float16)y1.y;
        a[14] = (_Float16)y1.z; a[15] = (_Float16)y1.w;
        afrag[kc] = a;
    }
}

__device__ __forceinline__ void mm_tiles_128(const v16h afrag[4],
                                             const _Float16* __restrict__ wimg,
                                             const float* __restrict__ bias,
                                             float* __restrict__ OUT,
                                             int rowBase, int mrow, int hs, int nrows)
{
    const bool fullTile = (rowBase + 16 <= nrows);   // wave-uniform fast path
    #pragma unroll
    for (int n = 0; n < 8; ++n) {
        const float bcol = bias[16 * n + mrow];
        v8f acc;
        #pragma unroll
        for (int r = 0; r < 8; ++r) acc[r] = bcol;

        const _Float16* col = wimg + (16 * n + mrow) * LDK;
        #pragma unroll
        for (int kc = 0; kc < 4; ++kc) {
            const _Float16* bp = col + 32 * kc + 8 * hs;    // K = 32kc+16(p>>3)+8hs+(p&7)
            v8h lo = *(const v8h*)(bp);          // ds_load_b128
            v8h hi = *(const v8h*)(bp + 16);     // ds_load_b128
            v16h bfrag = __builtin_shufflevector(lo, hi,
                0, 1, 2, 3, 4, 5, 6, 7, 8, 9, 10, 11, 12, 13, 14, 15);
            acc = __builtin_amdgcn_wmma_f32_16x16x32_f16(
                false, afrag[kc], false, bfrag, (short)0, acc, false, false);
        }
        // C/D layout: VGPR r holds M = r + 8*hs, N = mrow
        float* obase = OUT + (size_t)(rowBase + 8 * hs) * HIDDIM + 16 * n + mrow;
        if (fullTile) {
            #pragma unroll
            for (int r = 0; r < 8; ++r) obase[(size_t)r * HIDDIM] = acc[r];
        } else {
            #pragma unroll
            for (int r = 0; r < 8; ++r) {
                if (rowBase + 8 * hs + r < nrows) obase[(size_t)r * HIDDIM] = acc[r];
            }
        }
    }
}

// ---------------------------------------------------------------------------
// OUT = H @ W + b, single packed weight (used for the input projection)
// 256 threads = 8 waves -> 128 rows per block.
// ---------------------------------------------------------------------------
__global__ void __launch_bounds__(256)
gemm128_wmma(const float* __restrict__ H, const _Float16* __restrict__ Wp,
             const float* __restrict__ bias, float* __restrict__ OUT, int nrows)
{
    extern __shared__ _Float16 smem[];
    const int tid = threadIdx.x;
    async_stage(Wp, smem, WSLOT * 2, tid, 256);

    const int lane = tid & 31, wave = tid >> 5;
    const int hs = lane >> 4, mrow = lane & 15;
    const int rowBase = blockIdx.x * 128 + wave * 16;

    v16h afrag[4];
    load_afrag(H, rowBase, mrow, hs, nrows, afrag);
    mm_tiles_128(afrag, smem, bias, OUT, rowBase, mrow, hs, nrows);
}

// ---------------------------------------------------------------------------
// fused q/k/v/skip: 4 packed weights staged together (139 KB LDS),
// A fragments loaded once per wave and reused for all four products.
// 512 threads = 16 waves -> 256 rows per block.
// ---------------------------------------------------------------------------
__global__ void __launch_bounds__(512)
gemm128x4_wmma(const float* __restrict__ H, const _Float16* __restrict__ Wp4,
               const float* __restrict__ b0, const float* __restrict__ b1,
               const float* __restrict__ b2, const float* __restrict__ b3,
               float* __restrict__ o0, float* __restrict__ o1,
               float* __restrict__ o2, float* __restrict__ o3, int nrows)
{
    extern __shared__ _Float16 smem[];
    const int tid = threadIdx.x;
    async_stage(Wp4, smem, 4 * WSLOT * 2, tid, 512);

    const int lane = tid & 31, wave = tid >> 5;
    const int hs = lane >> 4, mrow = lane & 15;
    const int rowBase = blockIdx.x * 256 + wave * 16;

    v16h afrag[4];
    load_afrag(H, rowBase, mrow, hs, nrows, afrag);

    const float* biases[4] = { b0, b1, b2, b3 };
    float*       outs[4]   = { o0, o1, o2, o3 };
    #pragma unroll
    for (int m = 0; m < 4; ++m)
        mm_tiles_128(afrag, smem + m * WSLOT, biases[m], outs[m],
                     rowBase, mrow, hs, nrows);
}

// ---------------------------------------------------------------------------
// zero aggregation buffer / softmax sums, seed segment-max with -inf
// ---------------------------------------------------------------------------
__global__ void __launch_bounds__(256)
init_buffers(float* __restrict__ agg, float* __restrict__ ssum,
             int* __restrict__ mOrd, int n_nodes)
{
    int t = blockIdx.x * blockDim.x + threadIdx.x;
    if (t < n_nodes * HIDDIM) agg[t] = 0.0f;
    if (t < n_nodes * NHEADS) {
        ssum[t] = 0.0f;
        mOrd[t] = f2ord(-__builtin_huge_valf());
    }
}

// ---------------------------------------------------------------------------
// wave32 per edge: alpha[e,h] = scale * <q[dst,h,:], k[src,h,:]>; segment max
// ---------------------------------------------------------------------------
__global__ void __launch_bounds__(256)
edge_logits(const float* __restrict__ q, const float* __restrict__ k,
            const long long* __restrict__ ei, int* __restrict__ mOrd,
            float* __restrict__ alpha, int E)
{
    int e    = (blockIdx.x * blockDim.x + threadIdx.x) >> 5;
    int lane = threadIdx.x & 31;
    if (e >= E) return;
    const long long s = ei[e];
    const long long d = ei[(size_t)E + e];
    const int hh = lane >> 3;
    const int c4 = (lane & 7) * 4;
    const float4 qv = *(const float4*)(q + (size_t)d * HIDDIM + hh * CDIM + c4);
    const float4 kv = *(const float4*)(k + (size_t)s * HIDDIM + hh * CDIM + c4);
    float p = qv.x * kv.x + qv.y * kv.y + qv.z * kv.z + qv.w * kv.w;
    p += __shfl_xor(p, 1, 32);
    p += __shfl_xor(p, 2, 32);
    p += __shfl_xor(p, 4, 32);
    p *= 0.17677669529663687f;               // 1/sqrt(32)
    if ((lane & 7) == 0) {
        alpha[(size_t)e * NHEADS + hh] = p;
        atomicMax(mOrd + (size_t)d * NHEADS + hh, f2ord(p));
    }
}

// ---------------------------------------------------------------------------
// thread per (edge, head): a = exp(alpha - max); segment sum
// ---------------------------------------------------------------------------
__global__ void __launch_bounds__(256)
edge_exp(float* __restrict__ alpha, const int* __restrict__ mOrd,
         float* __restrict__ ssum, const long long* __restrict__ ei, int E)
{
    int t = blockIdx.x * blockDim.x + threadIdx.x;
    if (t >= E * NHEADS) return;
    int e  = t >> 2;
    int hh = t & 3;
    const long long d = ei[(size_t)E + e];
    float a = expf(alpha[t] - ord2f(mOrd[(size_t)d * NHEADS + hh]));
    alpha[t] = a;
    atomicAdd(ssum + (size_t)d * NHEADS + hh, a);
}

// ---------------------------------------------------------------------------
// wave32 per edge: agg[dst] += (a / (s+eps)) * v[src]
// ---------------------------------------------------------------------------
__global__ void __launch_bounds__(256)
edge_aggregate(const float* __restrict__ alpha, const float* __restrict__ ssum,
               const float* __restrict__ v, const long long* __restrict__ ei,
               float* __restrict__ agg, int E)
{
    int e    = (blockIdx.x * blockDim.x + threadIdx.x) >> 5;
    int lane = threadIdx.x & 31;
    if (e >= E) return;
    const long long s = ei[e];
    const long long d = ei[(size_t)E + e];
    const int hh = lane >> 3;
    const int c4 = (lane & 7) * 4;
    const float attn = alpha[(size_t)e * NHEADS + hh] /
                       (ssum[(size_t)d * NHEADS + hh] + 1e-16f);
    const float4 vv = *(const float4*)(v + (size_t)s * HIDDIM + hh * CDIM + c4);
    float* dst = agg + (size_t)d * HIDDIM + hh * CDIM + c4;
    atomicAdd(dst + 0, attn * vv.x);
    atomicAdd(dst + 1, attn * vv.y);
    atomicAdd(dst + 2, attn * vv.z);
    atomicAdd(dst + 3, attn * vv.w);
}

// ---------------------------------------------------------------------------
// wave32 per node: beta gate + residual + layernorm (in-place h update)
// beta logit = <o, w0+w2> + <xr, w1-w2>
// ---------------------------------------------------------------------------
__global__ void __launch_bounds__(256)
node_update(float* __restrict__ h, const float* __restrict__ agg,
            const float* __restrict__ xrbuf, const float* __restrict__ wbeta,
            const float* __restrict__ lng, const float* __restrict__ lnb,
            int n_nodes)
{
    int node = (blockIdx.x * blockDim.x + threadIdx.x) >> 5;
    int lane = threadIdx.x & 31;
    if (node >= n_nodes) return;
    const int idx = lane * 4;
    const size_t base = (size_t)node * HIDDIM + idx;

    const float4 o  = *(const float4*)(agg + base);
    const float4 xr = *(const float4*)(xrbuf + base);
    const float4 w0 = *(const float4*)(wbeta + idx);
    const float4 w1 = *(const float4*)(wbeta + 128 + idx);
    const float4 w2 = *(const float4*)(wbeta + 256 + idx);

    float part = o.x * (w0.x + w2.x) + o.y * (w0.y + w2.y)
               + o.z * (w0.z + w2.z) + o.w * (w0.w + w2.w)
               + xr.x * (w1.x - w2.x) + xr.y * (w1.y - w2.y)
               + xr.z * (w1.z - w2.z) + xr.w * (w1.w - w2.w);
    #pragma unroll
    for (int m = 1; m < 32; m <<= 1) part += __shfl_xor(part, m, 32);
    const float beta = 1.0f / (1.0f + expf(-part));
    const float ob   = 1.0f - beta;

    const float4 hv = *(const float4*)(h + base);
    float4 t;
    t.x = hv.x + beta * xr.x + ob * o.x;
    t.y = hv.y + beta * xr.y + ob * o.y;
    t.z = hv.z + beta * xr.z + ob * o.z;
    t.w = hv.w + beta * xr.w + ob * o.w;

    float sum = t.x + t.y + t.z + t.w;
    #pragma unroll
    for (int m = 1; m < 32; m <<= 1) sum += __shfl_xor(sum, m, 32);
    const float mean = sum * (1.0f / 128.0f);

    float dx = t.x - mean, dy = t.y - mean, dz = t.z - mean, dw = t.w - mean;
    float vsum = dx * dx + dy * dy + dz * dz + dw * dw;
    #pragma unroll
    for (int m = 1; m < 32; m <<= 1) vsum += __shfl_xor(vsum, m, 32);
    const float rstd = rsqrtf(vsum * (1.0f / 128.0f) + 1e-5f);

    const float4 gl = *(const float4*)(lng + idx);
    const float4 bl = *(const float4*)(lnb + idx);
    float4 out;
    out.x = dx * rstd * gl.x + bl.x;
    out.y = dy * rstd * gl.y + bl.y;
    out.z = dz * rstd * gl.z + bl.z;
    out.w = dw * rstd * gl.w + bl.w;
    *(float4*)(h + base) = out;
}

// ---------------------------------------------------------------------------
// wave32 per node: y[n] = <h[n], W_head> + b_head
// ---------------------------------------------------------------------------
__global__ void __launch_bounds__(256)
head_kernel(const float* __restrict__ h, const float* __restrict__ wh,
            const float* __restrict__ bh, float* __restrict__ y, int n_nodes)
{
    int node = (blockIdx.x * blockDim.x + threadIdx.x) >> 5;
    int lane = threadIdx.x & 31;
    if (node >= n_nodes) return;
    const int idx = lane * 4;
    const float4 hv = *(const float4*)(h + (size_t)node * HIDDIM + idx);
    const float4 w  = *(const float4*)(wh + idx);
    float part = hv.x * w.x + hv.y * w.y + hv.z * w.z + hv.w * w.w;
    #pragma unroll
    for (int m = 1; m < 32; m <<= 1) part += __shfl_xor(part, m, 32);
    if (lane == 0) y[node] = part + bh[0];
}

// ---------------------------------------------------------------------------
extern "C" void kernel_launch(void* const* d_in, const int* in_sizes, int n_in,
                              void* d_out, int out_size, void* d_ws, size_t ws_size,
                              hipStream_t stream)
{
    const float*     x_t   = (const float*)d_in[0];
    const long long* ei    = (const long long*)d_in[1];
    const float*     W_in  = (const float*)d_in[2];
    const float*     b_in  = (const float*)d_in[3];
    const float*     Wq    = (const float*)d_in[4];
    const float*     bq    = (const float*)d_in[5];
    const float*     Wk    = (const float*)d_in[6];
    const float*     bk    = (const float*)d_in[7];
    const float*     Wv    = (const float*)d_in[8];
    const float*     bv    = (const float*)d_in[9];
    const float*     Wsk   = (const float*)d_in[10];
    const float*     bsk   = (const float*)d_in[11];
    const float*     Wbeta = (const float*)d_in[12];
    const float*     lng   = (const float*)d_in[13];
    const float*     lnb   = (const float*)d_in[14];
    const float*     Wh    = (const float*)d_in[15];
    const float*     bh    = (const float*)d_in[16];
    float*           y     = (float*)d_out;

    const int N = in_sizes[0] / HIDDIM;   // 50000
    const int E = in_sizes[1] / 2;        // 800000
    const int L = in_sizes[5] / HIDDIM;   // 2

    // workspace layout (fp32 slots; every offset is a multiple of 4 -> 16B aligned)
    float* ws = (float*)d_ws;
    size_t off = 0;
    float* h     = ws + off; off += (size_t)N * HIDDIM;
    float* q     = ws + off; off += (size_t)N * HIDDIM;
    float* kbuf  = ws + off; off += (size_t)N * HIDDIM;
    float* vbuf  = ws + off; off += (size_t)N * HIDDIM;
    float* xr    = ws + off; off += (size_t)N * HIDDIM;
    float* agg   = ws + off; off += (size_t)N * HIDDIM;
    float* alpha = ws + off; off += (size_t)E * NHEADS;
    float* ssum  = ws + off; off += (size_t)N * NHEADS;
    int*   mOrd  = (int*)(ws + off); off += (size_t)N * NHEADS;
    _Float16* wpack = (_Float16*)(ws + off);   // (1 + 4L) packed images follow

    const dim3 blk(256);
    const int gPack   = (HIDDIM * HIDDIM + 255) / 256;       // 64
    const int gGemm   = (N + 127) / 128;
    const int gGemm4  = (N + 255) / 256;
    const int gInit   = (N * HIDDIM + 255) / 256;
    const int gEdgeW  = (int)(((size_t)E * 32 + 255) / 256);
    const int gEdgeEH = (E * NHEADS + 255) / 256;
    const int gNode   = (int)(((size_t)N * 32 + 255) / 256);

    // ---- pack all dense weights once into their LDS images ----
    pack_weight<<<gPack, blk, 0, stream>>>(W_in, wpack);
    for (int l = 0; l < L; ++l) {
        const size_t wo = (size_t)l * HIDDIM * HIDDIM;
        _Float16* base = wpack + (size_t)(1 + 4 * l) * WSLOT;
        pack_weight<<<gPack, blk, 0, stream>>>(Wq  + wo, base + 0 * (size_t)WSLOT);
        pack_weight<<<gPack, blk, 0, stream>>>(Wk  + wo, base + 1 * (size_t)WSLOT);
        pack_weight<<<gPack, blk, 0, stream>>>(Wv  + wo, base + 2 * (size_t)WSLOT);
        pack_weight<<<gPack, blk, 0, stream>>>(Wsk + wo, base + 3 * (size_t)WSLOT);
    }

    // ---- input projection ----
    gemm128_wmma<<<gGemm, blk, WSLOT * 2, stream>>>(x_t, wpack, b_in, h, N);

    for (int l = 0; l < L; ++l) {
        const size_t bo = (size_t)l * HIDDIM;
        const _Float16* w4 = wpack + (size_t)(1 + 4 * l) * WSLOT;

        gemm128x4_wmma<<<gGemm4, dim3(512), 4 * WSLOT * 2, stream>>>(
            h, w4, bq + bo, bk + bo, bv + bo, bsk + bo, q, kbuf, vbuf, xr, N);

        init_buffers<<<gInit, blk, 0, stream>>>(agg, ssum, mOrd, N);
        edge_logits<<<gEdgeW, blk, 0, stream>>>(q, kbuf, ei, mOrd, alpha, E);
        edge_exp<<<gEdgeEH, blk, 0, stream>>>(alpha, mOrd, ssum, ei, E);
        edge_aggregate<<<gEdgeW, blk, 0, stream>>>(alpha, ssum, vbuf, ei, agg, E);
        node_update<<<gNode, blk, 0, stream>>>(h, agg, xr,
                                               Wbeta + (size_t)l * 3 * HIDDIM,
                                               lng + bo, lnb + bo, N);
    }

    head_kernel<<<gNode, blk, 0, stream>>>(h, Wh, bh, y, N);
}
